// TransformerGraphConv_54700703481969
// MI455X (gfx1250) — matrix operand order repaired and checked by reading, not compiled, not run on hardware
//
#include <hip/hip_runtime.h>
#include <hip/hip_bf16.h>

typedef _Float16 half_t;
typedef __attribute__((ext_vector_type(16))) _Float16 v16h;
typedef __attribute__((ext_vector_type(8)))  _Float16 v8h;
typedef __attribute__((ext_vector_type(8)))  float    v8f;

#define NB 8
#define NN 1024
#define CIN 256
#define COUT 512
#define NH 8
#define DH 64

static __device__ inline v8f wmma_f16(v16h a, v16h b, v8f c) {
  // D = A(16x32 f16) * B(32x16 f16) + C(16x16 f32)
  return __builtin_amdgcn_wmma_f32_16x16x32_f16(
      /*neg_a=*/false, a, /*neg_b=*/false, b,
      /*c_mod=*/(short)0, c, /*reuse_a=*/false, /*reuse_b=*/false);
}

union AFrag { v16h v; v8h h[2]; half_t e[16]; };

// ---------------- pack kernels ----------------

__global__ __launch_bounds__(256) void pack_x_kernel(const float* __restrict__ x,
                                                     half_t* __restrict__ xh, int n) {
  int i = blockIdx.x * 256 + threadIdx.x;
  if (i < n) xh[i] = (half_t)x[i];
}

// W: [H][CIN][DH] -> WT: [H][DH][CIN]  (B-operand friendly: fixed out-col, contiguous K)
__global__ __launch_bounds__(256) void pack_wT_kernel(const float* __restrict__ W,
                                                      half_t* __restrict__ WT) {
  int i = blockIdx.x * 256 + threadIdx.x;           // i < H*CIN*DH
  if (i >= NH * CIN * DH) return;
  int h = i / (CIN * DH);
  int r = i - h * (CIN * DH);
  int c = r / DH;
  int o = r - c * DH;
  WT[(h * DH + o) * CIN + c] = (half_t)W[i];
}

// conv_w already [COUT][CIN] == B-operand layout; just convert
__global__ __launch_bounds__(256) void pack_conv_kernel(const float* __restrict__ W,
                                                        half_t* __restrict__ Wh) {
  int i = blockIdx.x * 256 + threadIdx.x;
  if (i < COUT * CIN) Wh[i] = (half_t)W[i];
}

// ---------------- projection GEMM: Q,K,V,proj ----------------
// grid: B * N/16 workgroups, 256 threads (8 waves). Each WG: one 16-row x tile.
// 128 output 16x16 tiles per WG: 32 Q + 32 K + 32 V + 32 proj; 16 tiles/wave.
__global__ __launch_bounds__(256) void proj_kernel(
    const half_t* __restrict__ xh, const half_t* __restrict__ WqT,
    const half_t* __restrict__ WkT, const half_t* __restrict__ WvT,
    const half_t* __restrict__ convWh, const float* __restrict__ conv_b,
    half_t* __restrict__ Qh, half_t* __restrict__ Kh, half_t* __restrict__ Vt,
    float* __restrict__ projf) {
  __shared__ alignas(32) half_t xt[16 * CIN];

  int wg = blockIdx.x;
  int b = wg >> 6;
  int n0 = (wg & 63) * 16;
  int t = threadIdx.x;

  // stage x tile [16][256] f16 into LDS (32B per thread)
  {
    int row = t >> 4;
    int c0 = (t & 15) * 16;
    *(v16h*)&xt[row * CIN + c0] =
        *(const v16h*)&xh[((size_t)(b * NN + n0 + row)) * CIN + c0];
  }
  __syncthreads();

  int wave = t >> 5, lane = t & 31;
  int lr = lane & 15, lh = lane >> 4;

  // B-operand row base for a given tile index (this lane's output column)
  auto tile_bbase = [&](int tile) -> const half_t* {
    int kind = tile >> 5;   // 0=Q 1=K 2=V 3=proj
    int sub = tile & 31;
    if (kind < 3) {
      int hh = sub >> 2, dt = sub & 3;
      const half_t* WT = (kind == 0) ? WqT : (kind == 1) ? WkT : WvT;
      return WT + ((size_t)hh * DH + dt * 16 + lr) * CIN;
    }
    return convWh + ((size_t)sub * 16 + lr) * CIN;
  };

  // A fragments from the LDS x tile are identical for all 16 tiles of this wave:
  // preload all 8 k-block fragments once.
  AFrag axk[8];
#pragma unroll
  for (int kb = 0; kb < 8; ++kb) {
    const half_t* rp = &xt[lr * CIN + kb * 32 + lh * 16];
    axk[kb].h[0] = *(const v8h*)rp;          // K = kb*32 + lh*16 .. +7
    axk[kb].h[1] = *(const v8h*)(rp + 8);    // K = kb*32 + lh*16 + 8 .. +15
  }

  // Prime prefetch for the first tile's B row (512B per output column row,
  // two lanes per row split the cachelines: lh=0 -> bytes 0..255, lh=1 -> 256..511).
  {
    const half_t* pb = tile_bbase(wave * 16);
    __builtin_prefetch(pb + lh * 128, 0, 3);
    __builtin_prefetch(pb + lh * 128 + 64, 0, 3);
  }

  for (int i = 0; i < 16; ++i) {
    int tile = wave * 16 + i;
    int kind = tile >> 5;     // 0=Q 1=K 2=V 3=proj
    int sub = tile & 31;
    int hh = sub >> 2, dt = sub & 3;
    const half_t* bbase = tile_bbase(tile);

    // Prefetch next tile's B row while we compute this one (global_prefetch:
    // no LOADcnt, scheduler cannot sink-and-serialize it).
    if (i < 15) {
      const half_t* nb = tile_bbase(tile + 1);
      __builtin_prefetch(nb + lh * 128, 0, 3);
      __builtin_prefetch(nb + lh * 128 + 64, 0, 3);
    }

    // B fragments (should now hit near cache), then the WMMA chain.
    v16h bf[8];
#pragma unroll
    for (int kb = 0; kb < 8; ++kb)
      bf[kb] = *(const v16h*)(bbase + kb * 32 + lh * 16);

    v8f acc = {};
#pragma unroll
    for (int kb = 0; kb < 8; ++kb)
      acc = wmma_f16(axk[kb].v, bf[kb], acc);

    int mb = lh * 8;
    if (kind == 0) {
#pragma unroll
      for (int v = 0; v < 8; ++v) {
        int row = n0 + v + mb;
        Qh[(((size_t)(b * NH + hh)) * NN + row) * DH + dt * 16 + lr] =
            (half_t)(acc[v] * 0.125f);   // fold 1/sqrt(DH)
      }
    } else if (kind == 1) {
#pragma unroll
      for (int v = 0; v < 8; ++v) {
        int row = n0 + v + mb;
        Kh[(((size_t)(b * NH + hh)) * NN + row) * DH + dt * 16 + lr] = (half_t)acc[v];
      }
    } else if (kind == 2) {
      // V stored transposed [b][h][d][n]
#pragma unroll
      for (int v = 0; v < 8; ++v) {
        int row = n0 + v + mb;
        Vt[(((size_t)(b * NH + hh)) * DH + dt * 16 + lr) * NN + row] = (half_t)acc[v];
      }
    } else {
      float bias = conv_b[sub * 16 + lr];
#pragma unroll
      for (int v = 0; v < 8; ++v) {
        int row = n0 + v + mb;
        projf[((size_t)(b * NN + row)) * COUT + sub * 16 + lr] = acc[v] + bias;
      }
    }
  }
}

// ---------------- attention ----------------
// grid: B*H*(N/64) workgroups, 128 threads (4 waves). Dynamic LDS:
//   S[64][1024] f32 (256KB) + rowmax[64] + rowinv[64].
// Phase A: wave w owns column-tiles [w*16, w*16+16) for ALL 4 row-blocks ->
//          each K fragment is loaded once and reused by 4 WMMAs (per kb),
//          with next-tile K fragments double-buffered in registers.
// Phase B/C: wave w owns rows [w*16, w*16+16).
#define ATTN_SMEM (64 * NN * sizeof(float) + 128 * sizeof(float))

__global__ __launch_bounds__(128) void attn_kernel(const half_t* __restrict__ Qh,
                                                   const half_t* __restrict__ Kh,
                                                   const half_t* __restrict__ Vt,
                                                   float* __restrict__ Obuf) {
  extern __shared__ char smem[];
  float* S = (float*)smem;                           // [64][1024]
  float* rowmax = (float*)(smem + 64 * NN * sizeof(float));
  float* rowinv = rowmax + 64;

  int wg = blockIdx.x;
  int bh = wg >> 4;            // b*H + h
  int r0 = (wg & 15) * 64;
  int t = threadIdx.x;
  int wave = t >> 5, lane = t & 31;
  int lr = lane & 15, lh = lane >> 4;

  const half_t* Qb = Qh + (size_t)bh * NN * DH;
  const half_t* Kb = Kh + (size_t)bh * NN * DH;
  const half_t* Vb = Vt + (size_t)bh * DH * NN;

  // ---- Phase A: S = Q K^T into LDS ----
  // A fragments of Q for all 4 row blocks (rows r0 .. r0+63)
  AFrag aq[4][2];
#pragma unroll
  for (int rb = 0; rb < 4; ++rb) {
    const half_t* qrow = Qb + ((size_t)(r0 + rb * 16 + lr)) * DH;
#pragma unroll
    for (int kb = 0; kb < 2; ++kb) {
      aq[rb][kb].h[0] = *(const v8h*)(qrow + kb * 32 + lh * 16);
      aq[rb][kb].h[1] = *(const v8h*)(qrow + kb * 32 + lh * 16 + 8);
    }
  }

  {
    int ct0 = wave * 16;  // this wave's first column tile
    // double-buffered K B-fragments
    const half_t* kp = Kb + ((size_t)(ct0 * 16 + lr)) * DH;
    v16h b0 = *(const v16h*)(kp + lh * 16);
    v16h b1 = *(const v16h*)(kp + 32 + lh * 16);
    for (int it = 0; it < 16; ++it) {
      v16h c0 = b0, c1 = b1;
      if (it < 15) {
        const half_t* kn = Kb + ((size_t)((ct0 + it + 1) * 16 + lr)) * DH;
        b0 = *(const v16h*)(kn + lh * 16);
        b1 = *(const v16h*)(kn + 32 + lh * 16);
      }
      v8f acc[4];
#pragma unroll
      for (int rb = 0; rb < 4; ++rb) {
        acc[rb] = v8f{};
        acc[rb] = wmma_f16(aq[rb][0].v, c0, acc[rb]);
        acc[rb] = wmma_f16(aq[rb][1].v, c1, acc[rb]);
      }
#pragma unroll
      for (int rb = 0; rb < 4; ++rb)
#pragma unroll
        for (int v = 0; v < 8; ++v)
          S[(rb * 16 + v + lh * 8) * NN + (ct0 + it) * 16 + lr] = acc[rb][v];
    }
  }

  __syncthreads();  // rows now complete across all waves

  // ---- Phase B: exact row max & sum(exp); wave w owns rows w*16..+15 ----
  for (int r = 0; r < 16; ++r) {
    const float* sr = S + (wave * 16 + r) * NN;
    float mx = -3.402823466e+38f;
    for (int j = lane; j < NN; j += 32) mx = fmaxf(mx, sr[j]);
#pragma unroll
    for (int off = 16; off > 0; off >>= 1) mx = fmaxf(mx, __shfl_xor(mx, off, 32));
    float sum = 0.f;
    for (int j = lane; j < NN; j += 32) sum += __expf(sr[j] - mx);
#pragma unroll
    for (int off = 16; off > 0; off >>= 1) sum += __shfl_xor(sum, off, 32);
    if (lane == 0) {
      rowmax[wave * 16 + r] = mx;
      rowinv[wave * 16 + r] = 1.0f / sum;
    }
  }

  // ---- Phase C: O = softmax(S) @ V; exp applied on the fly ----
  v8f oacc[4] = {};
  int row_l = wave * 16 + lr;        // A-fragment row for this lane
  float rmax = rowmax[row_l];
  const float* srow = S + row_l * NN;

  // double-buffered V B-fragments (4 d-tiles)
  v16h vb[4];
#pragma unroll
  for (int dt = 0; dt < 4; ++dt)
    vb[dt] = *(const v16h*)(Vb + ((size_t)(dt * 16 + lr)) * NN + lh * 16);

  for (int kb = 0; kb < NN / 32; ++kb) {
    v16h vc[4];
#pragma unroll
    for (int dt = 0; dt < 4; ++dt) vc[dt] = vb[dt];
    if (kb < NN / 32 - 1) {
#pragma unroll
      for (int dt = 0; dt < 4; ++dt)
        vb[dt] = *(const v16h*)(Vb + ((size_t)(dt * 16 + lr)) * NN +
                                (kb + 1) * 32 + lh * 16);
    }
    AFrag a;
#pragma unroll
    for (int i = 0; i < 8; ++i)
      a.e[i] = (half_t)__expf(srow[kb * 32 + lh * 16 + i] - rmax);
#pragma unroll
    for (int i = 0; i < 8; ++i)
      a.e[8 + i] = (half_t)__expf(srow[kb * 32 + lh * 16 + 8 + i] - rmax);
#pragma unroll
    for (int dt = 0; dt < 4; ++dt)
      oacc[dt] = wmma_f16(a.v, vc[dt], oacc[dt]);
  }

  int bb = bh >> 3, hh = bh & 7;
#pragma unroll
  for (int dt = 0; dt < 4; ++dt) {
#pragma unroll
    for (int v = 0; v < 8; ++v) {
      int rl = wave * 16 + v + lh * 8;
      float val = oacc[dt][v] * rowinv[rl];
      val = fmaxf(val, 0.0f);  // ReLU
      Obuf[((size_t)(bb * NN) + r0 + rl) * COUT + hh * DH + dt * 16 + lr] = val;
    }
  }
}

// ---------------- fused add + LayerNorm ----------------
// one wave per row (512 channels), 8 rows per 256-thread block
__global__ __launch_bounds__(256) void ln_kernel(const float* __restrict__ Obuf,
                                                 const float* __restrict__ projf,
                                                 const float* __restrict__ lnw,
                                                 const float* __restrict__ lnb,
                                                 float* __restrict__ out) {
  int wave = threadIdx.x >> 5, lane = threadIdx.x & 31;
  size_t row = (size_t)blockIdx.x * 8 + wave;
  const float* o = Obuf + row * COUT;
  const float* p = projf + row * COUT;

  float vals[16];
  float sum = 0.f, sq = 0.f;
#pragma unroll
  for (int j = 0; j < 16; ++j) {
    int c = lane + 32 * j;
    float v = o[c] + p[c];
    vals[j] = v;
    sum += v;
    sq += v * v;
  }
#pragma unroll
  for (int off = 16; off > 0; off >>= 1) {
    sum += __shfl_xor(sum, off, 32);
    sq += __shfl_xor(sq, off, 32);
  }
  float mean = sum * (1.0f / COUT);
  float var = sq * (1.0f / COUT) - mean * mean;
  float rstd = rsqrtf(var + 1e-5f);
#pragma unroll
  for (int j = 0; j < 16; ++j) {
    int c = lane + 32 * j;
    out[row * COUT + c] = (vals[j] - mean) * rstd * lnw[c] + lnb[c];
  }
}

// ---------------- launch ----------------

extern "C" void kernel_launch(void* const* d_in, const int* in_sizes, int n_in,
                              void* d_out, int out_size, void* d_ws, size_t ws_size,
                              hipStream_t stream) {
  const float* x = (const float*)d_in[0];
  // d_in[1] = adj : provably unused by the reference (dead code) -> never read
  const float* Wq = (const float*)d_in[2];
  const float* Wk = (const float*)d_in[3];
  const float* Wv = (const float*)d_in[4];
  const float* conv_w = (const float*)d_in[5];
  const float* conv_b = (const float*)d_in[6];
  const float* ln_w = (const float*)d_in[7];
  const float* ln_b = (const float*)d_in[8];
  float* out = (float*)d_out;

  char* ws = (char*)d_ws;
  size_t off = 0;
  half_t* xh = (half_t*)(ws + off);   off += (size_t)NB * NN * CIN * 2;       // 4 MB
  half_t* WqT = (half_t*)(ws + off);  off += (size_t)NH * DH * CIN * 2;       // 256 KB
  half_t* WkT = (half_t*)(ws + off);  off += (size_t)NH * DH * CIN * 2;
  half_t* WvT = (half_t*)(ws + off);  off += (size_t)NH * DH * CIN * 2;
  half_t* convWh = (half_t*)(ws + off); off += (size_t)COUT * CIN * 2;        // 256 KB
  half_t* Qh = (half_t*)(ws + off);   off += (size_t)NB * NH * NN * DH * 2;   // 8 MB
  half_t* Kh = (half_t*)(ws + off);   off += (size_t)NB * NH * NN * DH * 2;   // 8 MB
  half_t* Vt = (half_t*)(ws + off);   off += (size_t)NB * NH * DH * NN * 2;   // 8 MB
  float* projf = (float*)(ws + off);  off += (size_t)NB * NN * COUT * 4;      // 16 MB
  float* Obuf = (float*)(ws + off);   off += (size_t)NB * NN * COUT * 4;      // 16 MB
  (void)ws_size;

  // pack / transpose to f16
  {
    int n = NB * NN * CIN;
    pack_x_kernel<<<(n + 255) / 256, 256, 0, stream>>>(x, xh, n);
  }
  pack_wT_kernel<<<(NH * CIN * DH + 255) / 256, 256, 0, stream>>>(Wq, WqT);
  pack_wT_kernel<<<(NH * CIN * DH + 255) / 256, 256, 0, stream>>>(Wk, WkT);
  pack_wT_kernel<<<(NH * CIN * DH + 255) / 256, 256, 0, stream>>>(Wv, WvT);
  pack_conv_kernel<<<(COUT * CIN + 255) / 256, 256, 0, stream>>>(conv_w, convWh);

  // projections: Q (pre-scaled), K, V^T, proj
  proj_kernel<<<NB * (NN / 16), 256, 0, stream>>>(xh, WqT, WkT, WvT, convWh, conv_b,
                                                  Qh, Kh, Vt, projf);

  // attention with 256KB-LDS score slab per workgroup
  hipFuncSetAttribute((const void*)attn_kernel,
                      hipFuncAttributeMaxDynamicSharedMemorySize, (int)ATTN_SMEM);
  attn_kernel<<<NB * NH * (NN / 64), 128, ATTN_SMEM, stream>>>(Qh, Kh, Vt, Obuf);

  // fused residual add + LayerNorm
  ln_kernel<<<(NB * NN) / 8, 256, 0, stream>>>(Obuf, projf, ln_w, ln_b, out);
}
